// SeqAE_89258010345574
// MI455X (gfx1250) — matrix operand order repaired and checked by reading, not compiled, not run on hardware
//
#include <hip/hip_runtime.h>

// Problem dims
#define Hn   2048              // n_hidden
#define In   512               // n_input / n_output
#define Gn   6144              // 3*Hn (gates r,z,n)
#define SEQn 2048
#define KSP  204               // int(2048*0.1)

// Persistent kernel config
#define NB    32               // persistent blocks (must all be co-resident)
#define PT    512              // threads per persistent block (16 wave32s)
#define NWAVE (NB * (PT / 32)) // 512 waves

typedef __attribute__((ext_vector_type(16))) __bf16 v16bf;
typedef __attribute__((ext_vector_type(8)))  __bf16 v8bf;
typedef __attribute__((ext_vector_type(8)))  float  v8f;

__device__ __forceinline__ unsigned short f2bf(float f) {
  unsigned u = __float_as_uint(f);
  u += 0x7FFFu + ((u >> 16) & 1u);   // round-to-nearest-even
  return (unsigned short)(u >> 16);
}

// ---------------------------------------------------------------------------
// WMMA mat-vec tile: y[row0..row0+16) = W_bf16[row0..row0+16, 0..K) @ h_bf16
// B matrix = h broadcast to all 16 columns -> every D column equals y tile.
// Layouts per CDNA5 ISA §7.12.2 (16-bit A 16x32, 16-bit B 32x16).
// ---------------------------------------------------------------------------
__device__ __forceinline__ void mv_tile(const unsigned short* __restrict__ W,
                                        const unsigned short* __restrict__ hv,
                                        float* __restrict__ y, int tile, int K) {
  const int lane = threadIdx.x & 31;
  const int m    = lane & 15;        // row within tile (A), col (B/D)
  const int hi   = lane >> 4;        // K-half selector
  const int row0 = tile << 4;
  const unsigned short* wrow = W + (size_t)(row0 + m) * (size_t)K + (size_t)(hi * 8);
  const unsigned short* hp   = hv + hi * 8;
  v8f acc = {};
  for (int k = 0; k < K; k += 32) {
    __builtin_prefetch(wrow + k + 512, 0, 1);                       // global_prefetch_b8
    v8bf a0 = *(const v8bf*)(const void*)(wrow + k);                // K = k+hi*8 .. +8
    v8bf a1 = *(const v8bf*)(const void*)(wrow + k + 16);           // K = k+16+hi*8 .. +8
    v16bf a = __builtin_shufflevector(a0, a1, 0,1,2,3,4,5,6,7,8,9,10,11,12,13,14,15);
    v8bf b0 = *(const v8bf*)(const void*)(hp + k);                  // h[k + hi*8 .. +8]
    v8bf b1 = *(const v8bf*)(const void*)(hp + k + 16);             // h[k + 16 + hi*8 .. +8]
    v16bf b = __builtin_shufflevector(b0, b1, 0,1,2,3,4,5,6,7,8,9,10,11,12,13,14,15);
    acc = __builtin_amdgcn_wmma_f32_16x16x32_bf16(false, a, false, b,
                                                  (short)0, acc, false, false);
  }
  // D layout: lane 0 holds (M=0..7, N=0) in vgprs 0..7; lane 16 holds M=8..15.
  if (m == 0) {
    float* yp = y + row0 + hi * 8;
#pragma unroll
    for (int g = 0; g < 8; ++g) yp[g] = acc[g];
  }
}

// ---------------------------------------------------------------------------
// Precompute Gi0[t, r] = (enc_Wih0 @ x_t)[r] for all t as one WMMA GEMM.
// A = X[tTile(16) x k(32)], B = W^T chunk (W row-major [Gn x In]).
// ---------------------------------------------------------------------------
__global__ void __launch_bounds__(256) gemm_gi0(const unsigned short* __restrict__ X,
                                                const unsigned short* __restrict__ W,
                                                float* __restrict__ Gi0) {
  const int wave = (blockIdx.x * 256 + threadIdx.x) >> 5;
  const int lane = threadIdx.x & 31;
  const int tilesR = Gn / 16;                  // 384
  const int t0 = (wave / tilesR) * 16;
  const int r0 = (wave % tilesR) * 16;
  const int m  = lane & 15;
  const int hi = lane >> 4;
  const unsigned short* xrow = X + (size_t)(t0 + m) * In + hi * 8;
  const unsigned short* wrow = W + (size_t)(r0 + m) * In + hi * 8;
  v8f acc = {};
  for (int k = 0; k < In; k += 32) {
    v8bf a0 = *(const v8bf*)(const void*)(xrow + k);
    v8bf a1 = *(const v8bf*)(const void*)(xrow + k + 16);
    v16bf a = __builtin_shufflevector(a0, a1, 0,1,2,3,4,5,6,7,8,9,10,11,12,13,14,15);
    v8bf b0 = *(const v8bf*)(const void*)(wrow + k);
    v8bf b1 = *(const v8bf*)(const void*)(wrow + k + 16);
    v16bf b = __builtin_shufflevector(b0, b1, 0,1,2,3,4,5,6,7,8,9,10,11,12,13,14,15);   // B[k+hi*8..+8, k+16+hi*8..+8; r0+m]
    acc = __builtin_amdgcn_wmma_f32_16x16x32_bf16(false, a, false, b,
                                                  (short)0, acc, false, false);
  }
  // D: lane holds column r0+m, rows t0 + hi*8 + g
  float* gp = Gi0 + (size_t)(t0 + hi * 8) * Gn + (r0 + m);
#pragma unroll
  for (int g = 0; g < 8; ++g) gp[(size_t)g * Gn] = acc[g];
}

// ---------------------------------------------------------------------------
__global__ void cvt_bf16(const float* __restrict__ s, unsigned short* __restrict__ d, int n) {
  int i = blockIdx.x * blockDim.x + threadIdx.x;
  int stride = gridDim.x * blockDim.x;
  for (; i < n; i += stride) d[i] = f2bf(s[i]);
}

__global__ void init_state(unsigned* __restrict__ ctrl, float* __restrict__ h0,
                           float* __restrict__ h1, unsigned short* __restrict__ h0b,
                           unsigned short* __restrict__ h1b) {
  int i = blockIdx.x * blockDim.x + threadIdx.x;
  int stride = gridDim.x * blockDim.x;
  if (i < 256) ctrl[i] = 0u;           // barrier counter + threshold + loss accum
  for (int j = i; j < Hn; j += stride) { h0[j] = 0.f; h1[j] = 0.f; h0b[j] = 0; h1b[j] = 0; }
}

// ---------------------------------------------------------------------------
struct RecArgs {
  unsigned* ctrl;                 // [0]=barrier counter; f[8]=topk thr; f[9]=loss accum
  float *h0, *h1, *ghb, *gib, *outb;
  unsigned short *h0b, *h1b, *xb;
  const float* Gi0;
  const unsigned short *eWhh0, *eWih1, *eWhh1;
  const unsigned short *dWih0, *dWhh0, *dWih1, *dWhh1, *dWout;
  const float *seq, *e_bih, *e_bhh, *d_bih, *d_bhh, *d_bout;
  float* out;
};

__device__ __forceinline__ void gru_update(int i, const float* __restrict__ gi,
                                           const float* __restrict__ gh,
                                           const float* __restrict__ bih,
                                           const float* __restrict__ bhh,
                                           float* __restrict__ h,
                                           unsigned short* __restrict__ hb) {
  float gir = gi[i]          + bih[i];
  float giz = gi[Hn + i]     + bih[Hn + i];
  float gin = gi[2 * Hn + i] + bih[2 * Hn + i];
  float ghr = gh[i]          + bhh[i];
  float ghz = gh[Hn + i]     + bhh[Hn + i];
  float ghn = gh[2 * Hn + i] + bhh[2 * Hn + i];
  float r = 1.f / (1.f + __expf(-(gir + ghr)));
  float z = 1.f / (1.f + __expf(-(giz + ghz)));
  float n = tanhf(gin + r * ghn);
  float hv = (1.f - z) * n + z * h[i];
  h[i]  = hv;
  hb[i] = f2bf(hv);
}

__global__ void __launch_bounds__(PT, 1) recurrence(RecArgs A) {
  __shared__ float s[Hn];
  const int tid   = threadIdx.x;
  const int gtid  = blockIdx.x * PT + tid;
  const int gwave = gtid >> 5;
  unsigned bidx = 0;

  auto bar = [&]() {
    __syncthreads();
    ++bidx;
    if (tid == 0) {
      __builtin_amdgcn_fence(__ATOMIC_RELEASE, "agent");
      __hip_atomic_fetch_add(A.ctrl, 1u, __ATOMIC_RELAXED, __HIP_MEMORY_SCOPE_AGENT);
      const unsigned target = bidx * (unsigned)NB;
      while (__hip_atomic_load(A.ctrl, __ATOMIC_RELAXED, __HIP_MEMORY_SCOPE_AGENT) < target)
        __builtin_amdgcn_s_sleep(1);
    }
    __syncthreads();
    __builtin_amdgcn_fence(__ATOMIC_ACQUIRE, "agent");   // invalidate stale WGP$/L0
  };

  // ======================= encoder: 2048 steps =======================
  for (int t = 0; t < SEQn; ++t) {
    // gh0 = enc_Whh0 @ h0  (384 tiles)
    if (gwave < Gn / 16) mv_tile(A.eWhh0, A.h0b, A.ghb, gwave, Hn);
    bar();
    if (gtid < Hn)
      gru_update(gtid, A.Gi0 + (size_t)t * Gn, A.ghb, A.e_bih, A.e_bhh, A.h0, A.h0b);
    bar();
    // gi1 = enc_Wih1 @ h0 ; gh1 = enc_Whh1 @ h1   (768 tiles)
    for (int tile = gwave; tile < 2 * (Gn / 16); tile += NWAVE) {
      if (tile < Gn / 16) mv_tile(A.eWih1, A.h0b, A.gib, tile, Hn);
      else                mv_tile(A.eWhh1, A.h1b, A.ghb, tile - Gn / 16, Hn);
    }
    bar();
    if (gtid < Hn)
      gru_update(gtid, A.gib, A.ghb, A.e_bih + Gn, A.e_bhh + Gn, A.h1, A.h1b);
    bar();
  }

  // ======================= top-k sparsity on h1 =======================
  if (blockIdx.x == 0) {
    for (int j = tid; j < Hn; j += PT) s[j] = A.h1[j];
    __syncthreads();
    for (int k = 2; k <= Hn; k <<= 1)
      for (int j = k >> 1; j > 0; j >>= 1) {
        for (int idx = tid; idx < Hn; idx += PT) {
          int ixj = idx ^ j;
          if (ixj > idx) {
            bool desc = ((idx & k) == 0);
            float a = s[idx], b = s[ixj];
            if (desc ? (a < b) : (a > b)) { s[idx] = b; s[ixj] = a; }
          }
        }
        __syncthreads();
      }
    if (tid == 0) ((float*)A.ctrl)[8] = s[KSP - 1];   // 204th largest
  }
  bar();
  {
    float thr = ((float*)A.ctrl)[8];
    if (gtid < Hn) {
      float v = A.h1[gtid];
      if (v < thr) v = 0.f;
      A.h1[gtid]  = v;
      A.h1b[gtid] = f2bf(v);
    }
    if (gtid < In) A.xb[gtid] = f2bf(A.seq[(size_t)(SEQn - 1) * In + gtid]);
  }
  bar();

  // ======================= decoder: 2047 steps =======================
  for (int st = 0; st < SEQn - 1; ++st) {
    // gi0 = dec_Wih0 @ x (K=512) ; gh0 = dec_Whh0 @ h0   (768 tiles)
    for (int tile = gwave; tile < 2 * (Gn / 16); tile += NWAVE) {
      if (tile < Gn / 16) mv_tile(A.dWih0, A.xb, A.gib, tile, In);
      else                mv_tile(A.dWhh0, A.h0b, A.ghb, tile - Gn / 16, Hn);
    }
    bar();
    if (gtid < Hn)
      gru_update(gtid, A.gib, A.ghb, A.d_bih, A.d_bhh, A.h0, A.h0b);
    bar();
    for (int tile = gwave; tile < 2 * (Gn / 16); tile += NWAVE) {
      if (tile < Gn / 16) mv_tile(A.dWih1, A.h0b, A.gib, tile, Hn);
      else                mv_tile(A.dWhh1, A.h1b, A.ghb, tile - Gn / 16, Hn);
    }
    bar();
    if (gtid < Hn)
      gru_update(gtid, A.gib, A.ghb, A.d_bih + Gn, A.d_bhh + Gn, A.h1, A.h1b);
    bar();
    // out = dec_Wout @ h1  (32 tiles)
    if (gwave < In / 16) mv_tile(A.dWout, A.h1b, A.outb, gwave, Hn);
    bar();
    // output + loss + next-step input (block 0, deterministic LDS reduce)
    if (blockIdx.x == 0) {
      float d2 = 0.f;
      if (tid < In) {
        float o = A.outb[tid] + A.d_bout[tid];
        A.out[(size_t)st * In + tid] = o;
        A.xb[tid] = f2bf(o);
        float df = o - A.seq[(size_t)(SEQn - 2 - st) * In + tid];
        d2 = df * df;
      }
      s[tid] = d2;
      __syncthreads();
      for (int w = PT / 2; w > 0; w >>= 1) {
        if (tid < w) s[tid] += s[tid + w];
        __syncthreads();
      }
      if (tid == 0) ((float*)A.ctrl)[9] += s[0] / (float)In;
    }
    bar();
  }
  if (blockIdx.x == 0 && tid == 0)
    A.out[(size_t)(SEQn - 1) * In] = ((float*)A.ctrl)[9] / (float)SEQn;
}

// ---------------------------------------------------------------------------
extern "C" void kernel_launch(void* const* d_in, const int* in_sizes, int n_in,
                              void* d_out, int out_size, void* d_ws, size_t ws_size,
                              hipStream_t stream) {
  (void)in_sizes; (void)n_in; (void)out_size; (void)ws_size;
  const float* seq   = (const float*)d_in[0];
  const float* eWih0 = (const float*)d_in[1];
  const float* eWih1 = (const float*)d_in[2];
  const float* eWhh  = (const float*)d_in[3];
  const float* e_bih = (const float*)d_in[4];
  const float* e_bhh = (const float*)d_in[5];
  const float* dWih0 = (const float*)d_in[6];
  const float* dWih1 = (const float*)d_in[7];
  const float* dWhh  = (const float*)d_in[8];
  const float* d_bih = (const float*)d_in[9];
  const float* d_bhh = (const float*)d_in[10];
  const float* dWout = (const float*)d_in[11];
  const float* d_bout= (const float*)d_in[12];
  float* out = (float*)d_out;

  char* base = (char*)d_ws;
  size_t off = 0;
  auto alloc = [&](size_t bytes) -> void* {
    void* p = base + off;
    off = (off + bytes + 255) & ~(size_t)255;
    return p;
  };
  unsigned*       ctrl = (unsigned*)alloc(1024);
  float*          h0   = (float*)alloc(Hn * 4);
  float*          h1   = (float*)alloc(Hn * 4);
  float*          ghb  = (float*)alloc(Gn * 4);
  float*          gib  = (float*)alloc(Gn * 4);
  float*          outb = (float*)alloc(In * 4);
  unsigned short* h0b  = (unsigned short*)alloc(Hn * 2);
  unsigned short* h1b  = (unsigned short*)alloc(Hn * 2);
  unsigned short* xb   = (unsigned short*)alloc(In * 2);
  unsigned short* seqb = (unsigned short*)alloc((size_t)SEQn * In * 2);
  float*          Gi0  = (float*)alloc((size_t)SEQn * Gn * 4);
  unsigned short* eWih0b = (unsigned short*)alloc((size_t)Gn * In * 2);
  unsigned short* eWhh0b = (unsigned short*)alloc((size_t)Gn * Hn * 2);
  unsigned short* eWih1b = (unsigned short*)alloc((size_t)Gn * Hn * 2);
  unsigned short* eWhh1b = (unsigned short*)alloc((size_t)Gn * Hn * 2);
  unsigned short* dWih0b = (unsigned short*)alloc((size_t)Gn * In * 2);
  unsigned short* dWhh0b = (unsigned short*)alloc((size_t)Gn * Hn * 2);
  unsigned short* dWih1b = (unsigned short*)alloc((size_t)Gn * Hn * 2);
  unsigned short* dWhh1b = (unsigned short*)alloc((size_t)Gn * Hn * 2);
  unsigned short* dWoutb = (unsigned short*)alloc((size_t)In * Hn * 2);

  init_state<<<8, 256, 0, stream>>>(ctrl, h0, h1, h0b, h1b);

  cvt_bf16<<<2048, 256, 0, stream>>>(seq,   seqb,   SEQn * In);
  cvt_bf16<<<2048, 256, 0, stream>>>(eWih0, eWih0b, Gn * In);
  cvt_bf16<<<2048, 256, 0, stream>>>(eWhh,                     eWhh0b, Gn * Hn);
  cvt_bf16<<<2048, 256, 0, stream>>>(eWhh + (size_t)Gn * Hn,   eWhh1b, Gn * Hn);
  cvt_bf16<<<2048, 256, 0, stream>>>(eWih1, eWih1b, Gn * Hn);
  cvt_bf16<<<2048, 256, 0, stream>>>(dWih0, dWih0b, Gn * In);
  cvt_bf16<<<2048, 256, 0, stream>>>(dWhh,                     dWhh0b, Gn * Hn);
  cvt_bf16<<<2048, 256, 0, stream>>>(dWhh + (size_t)Gn * Hn,   dWhh1b, Gn * Hn);
  cvt_bf16<<<2048, 256, 0, stream>>>(dWih1, dWih1b, Gn * Hn);
  cvt_bf16<<<2048, 256, 0, stream>>>(dWout, dWoutb, In * Hn);

  // All-timestep input-gate GEMM: (SEQ/16)*(Gn/16) = 49152 wave-tiles, 8 waves/block
  gemm_gi0<<<(SEQn / 16) * (Gn / 16) / 8, 256, 0, stream>>>(seqb, eWih0b, Gi0);

  RecArgs A;
  A.ctrl = ctrl;
  A.h0 = h0; A.h1 = h1; A.ghb = ghb; A.gib = gib; A.outb = outb;
  A.h0b = h0b; A.h1b = h1b; A.xb = xb;
  A.Gi0 = Gi0;
  A.eWhh0 = eWhh0b; A.eWih1 = eWih1b; A.eWhh1 = eWhh1b;
  A.dWih0 = dWih0b; A.dWhh0 = dWhh0b; A.dWih1 = dWih1b; A.dWhh1 = dWhh1b; A.dWout = dWoutb;
  A.seq = seq; A.e_bih = e_bih; A.e_bhh = e_bhh;
  A.d_bih = d_bih; A.d_bhh = d_bhh; A.d_bout = d_bout;
  A.out = out;

  recurrence<<<NB, PT, 0, stream>>>(A);
}
